// MultiHeadAttention_70300024701679
// MI455X (gfx1250) — matrix operand order repaired and checked by reading, not compiled, not run on hardware
//
#include <hip/hip_runtime.h>

typedef __bf16 bf16_t;
typedef __attribute__((ext_vector_type(16))) __bf16 v16bf;
typedef __attribute__((ext_vector_type(8)))  float  v8f;

#define B_DIM   4
#define S_DIM   2048
#define D_MODEL 1024
#define N_HEAD  16
#define DKV     64

__device__ __forceinline__ unsigned short f2bf_bits(float f) {
  union { float f; unsigned u; } x; x.f = f;
  unsigned r = x.u + 0x7FFFu + ((x.u >> 16) & 1u);   // round-to-nearest-even
  return (unsigned short)(r >> 16);
}
__device__ __forceinline__ bf16_t f2bf(float f) {
  unsigned short h = f2bf_bits(f);
  bf16_t out;
  __builtin_memcpy(&out, &h, 2);
  return out;
}

__device__ __forceinline__ v8f wmma_bf16(v16bf a, v16bf b, v8f c) {
  return __builtin_amdgcn_wmma_f32_16x16x32_bf16(false, a, false, b, (short)0, c,
                                                 false, false);
}

// gfx1250 async copy: 16B per lane, global -> LDS, tracked by ASYNCcnt.
__device__ __forceinline__ void async_copy_b128(bf16_t* lds_dst,
                                                const bf16_t* gsrc) {
  unsigned loff = (unsigned)(uintptr_t)lds_dst;  // low 32 bits = LDS offset
  asm volatile("global_load_async_to_lds_b128 %0, %1, off"
               :
               : "v"(loff), "v"(gsrc)
               : "memory");
}
__device__ __forceinline__ void wait_async() {
  asm volatile("s_wait_asynccnt 0x0" ::: "memory");
}

// A-matrix 16x32 bf16 fragment, row-major source, leading dim `ld` (elems).
// lanes 0-15 (M=lane): elems 0..7 = K0..7, 8..15 = K16..23
// lanes 16-31 (M=lane-16): elems 0..7 = K8..15, 8..15 = K24..31
__device__ __forceinline__ v16bf load_a_frag(const bf16_t* base, int ld) {
  int lane = threadIdx.x & 31;
  int r  = lane & 15;
  int hi = lane >> 4;
  const bf16_t* p = base + (size_t)r * ld + hi * 8;
  v16bf f;
#pragma unroll
  for (int i = 0; i < 8; ++i) f[i] = p[i];
#pragma unroll
  for (int i = 0; i < 8; ++i) f[8 + i] = p[16 + i];
  return f;
}

// B-matrix 32x16 bf16 fragment from K-contiguous storage (row n = 32 K-values
// of output column n), leading dim `ld` (elems).
// lanes 0-15 (N=lane): elems 0..7 = K0..7, 8..15 = K16..23 ; lanes 16-31: K8..15, K24..31
__device__ __forceinline__ v16bf load_b_frag(const bf16_t* base, int ld) {
  int lane = threadIdx.x & 31;
  int n  = lane & 15;
  int hi = lane >> 4;
  const bf16_t* p = base + (size_t)n * ld + hi * 8;
  v16bf f;
#pragma unroll
  for (int i = 0; i < 8; ++i) f[i] = p[i];
#pragma unroll
  for (int i = 0; i < 8; ++i) f[8 + i] = p[16 + i];
  return f;
}

// ---------------------------------------------------------------------------
// Prepass A: fp32 -> bf16 elementwise for the three X inputs.
// ---------------------------------------------------------------------------
__global__ __launch_bounds__(256) void convert_x_kernel(
    const float* __restrict__ Xq, const float* __restrict__ Xk,
    const float* __restrict__ Xv, bf16_t* __restrict__ Oq,
    bf16_t* __restrict__ Ok, bf16_t* __restrict__ Ov) {
  const int z = blockIdx.y;
  const float* X = (z == 0) ? Xq : (z == 1) ? Xk : Xv;
  bf16_t* O = (z == 0) ? Oq : (z == 1) ? Ok : Ov;
  size_t i = ((size_t)blockIdx.x * 256 + threadIdx.x) * 4;
  float4 v = *(const float4*)(X + i);
  unsigned lo = (unsigned)f2bf_bits(v.x) | ((unsigned)f2bf_bits(v.y) << 16);
  unsigned hi = (unsigned)f2bf_bits(v.z) | ((unsigned)f2bf_bits(v.w) << 16);
  uint2 pk; pk.x = lo; pk.y = hi;
  *(uint2*)(O + i) = pk;
}

// ---------------------------------------------------------------------------
// Prepass B: fp32 W [K][N] -> bf16 W^T [N][K] (LDS tile transpose).
// ---------------------------------------------------------------------------
__global__ __launch_bounds__(256) void convert_w_kernel(
    const float* __restrict__ Wq, const float* __restrict__ Wk,
    const float* __restrict__ Wv, bf16_t* __restrict__ Tq,
    bf16_t* __restrict__ Tk, bf16_t* __restrict__ Tv) {
  __shared__ float tile[32][33];
  const int z = blockIdx.z;
  const float* W = (z == 0) ? Wq : (z == 1) ? Wk : Wv;
  bf16_t* T = (z == 0) ? Tq : (z == 1) ? Tk : Tv;
  const int n0 = blockIdx.x * 32, k0 = blockIdx.y * 32;
  const int tx = threadIdx.x & 31, ty = threadIdx.x >> 5;  // 32 x 8
#pragma unroll
  for (int j = 0; j < 32; j += 8)
    tile[ty + j][tx] = W[(size_t)(k0 + ty + j) * D_MODEL + n0 + tx];
  __syncthreads();
#pragma unroll
  for (int j = 0; j < 32; j += 8)
    T[(size_t)(n0 + ty + j) * D_MODEL + k0 + tx] = f2bf(tile[tx][ty + j]);
}

// ---------------------------------------------------------------------------
// Kernel 1: QKV projection GEMM, bf16 operands staged into double-buffered
// LDS with global_load_async_to_lds_b128 so the tile-(k+1) fetch overlaps the
// tile-k WMMAs. Block tile 128x64, K-step 32; 8 waves each own one 16-row M
// subtile x four 16-col N subtiles -> 4 WMMA per K-step.
// Q,K stored bf16 [B,H,S,64]; V stored transposed bf16 [B,H,64,S].
// ---------------------------------------------------------------------------
#define GBM 128
#define GBN 64
#define GBK 32
#define LDP 40   // padded LDS stride (bf16 elems); 80B rows, 16B aligned

__device__ __forceinline__ void stage_tiles(const bf16_t* __restrict__ X,
                                            const bf16_t* __restrict__ W,
                                            bf16_t* As, bf16_t* Bs, int m0,
                                            int n0, int k0, int tid) {
  // A tile: 128x32 bf16 = 512 chunks of 16B; 2 async b128 per thread
#pragma unroll
  for (int i = 0; i < 2; ++i) {
    int chunk = tid + i * 256;
    int r = chunk >> 2, c8 = (chunk & 3) * 8;
    async_copy_b128(&As[r * LDP + c8],
                    X + (size_t)(m0 + r) * D_MODEL + k0 + c8);
  }
  // B tile: 64x32 bf16 = 256 chunks of 16B; 1 async b128 per thread
  int n = tid >> 2, c8 = (tid & 3) * 8;
  async_copy_b128(&Bs[n * LDP + c8],
                  W + (size_t)(n0 + n) * D_MODEL + k0 + c8);
}

__global__ __launch_bounds__(256) void qkv_proj_kernel(
    const bf16_t* __restrict__ Xq, const bf16_t* __restrict__ Xk,
    const bf16_t* __restrict__ Xv, const bf16_t* __restrict__ Wq,
    const bf16_t* __restrict__ Wk, const bf16_t* __restrict__ Wv,
    bf16_t* __restrict__ Qo, bf16_t* __restrict__ Ko, bf16_t* __restrict__ Vo) {
  __shared__ bf16_t As[2][GBM * LDP];   // double-buffered X tile (M x K)
  __shared__ bf16_t Bs[2][GBN * LDP];   // double-buffered W^T tile (N x K)

  const int z = blockIdx.z;
  const bf16_t* X = (z == 0) ? Xq : (z == 1) ? Xk : Xv;
  const bf16_t* W = (z == 0) ? Wq : (z == 1) ? Wk : Wv;

  const int m0 = blockIdx.x * GBM;           // over 8192
  const int n0 = blockIdx.y * GBN;           // over 1024
  const int tid  = threadIdx.x;
  const int wave = tid >> 5;                 // M subtile 0..7
  const int lane = tid & 31;

  v8f acc[4] = {{}, {}, {}, {}};

  // prologue: stage tile 0 into buffer 0
  stage_tiles(X, W, As[0], Bs[0], m0, n0, 0, tid);
  wait_async();
  __syncthreads();

  for (int k0 = 0, it = 0; k0 < D_MODEL; k0 += GBK, ++it) {
    const int cur = it & 1;
    // kick off the next tile's async copies into the other buffer
    if (k0 + GBK < D_MODEL)
      stage_tiles(X, W, As[cur ^ 1], Bs[cur ^ 1], m0, n0, k0 + GBK, tid);

    // compute on the current buffer while the copies are in flight
    v16bf a = load_a_frag(&As[cur][(wave * 16) * LDP], LDP);
#pragma unroll
    for (int nt = 0; nt < 4; ++nt) {
      v16bf b = load_b_frag(&Bs[cur][(nt * 16) * LDP], LDP);
      acc[nt] = wmma_bf16(a, b, acc[nt]);
    }

    wait_async();
    __syncthreads();
  }

  bf16_t* Out = (z == 0) ? Qo : (z == 1) ? Ko : Vo;
  const int col = lane & 15;
  const int rbase = (lane >> 4) * 8;
#pragma unroll
  for (int nt = 0; nt < 4; ++nt) {
    int n = n0 + nt * 16 + col;
    int h = n >> 6, d = n & 63;
#pragma unroll
    for (int g = 0; g < 8; ++g) {
      int m = m0 + wave * 16 + rbase + g;
      int b = m >> 11;                        // / 2048
      int s = m & 2047;
      size_t addr;
      if (z == 2)  // V transposed: [B,H,64,S]
        addr = ((size_t)(b * N_HEAD + h) * DKV + d) * S_DIM + s;
      else         // Q,K: [B,H,S,64]
        addr = ((size_t)(b * N_HEAD + h) * S_DIM + s) * DKV + d;
      Out[addr] = f2bf(acc[nt][g]);
    }
  }
}

// ---------------------------------------------------------------------------
// Kernel 2: attention. One workgroup per (b*H+h, 16-row q tile).
// Dynamic LDS: scores f32 [16][2064] | P bf16 [16][2064] | ctx f32 [16][64]
// ---------------------------------------------------------------------------
#define SSTR (S_DIM + 16)
#define ATTN_SMEM_BYTES (16 * SSTR * 4 + 16 * SSTR * 2 + 16 * DKV * 4)

__global__ __launch_bounds__(256) void attention_kernel(
    const bf16_t* __restrict__ Q, const bf16_t* __restrict__ K,
    const bf16_t* __restrict__ Vt, float* __restrict__ ctx_out,
    float* __restrict__ attn_out) {
  extern __shared__ char smem_raw[];
  float*  Ssc = (float*)smem_raw;                              // [16][SSTR]
  bf16_t* Pb  = (bf16_t*)(smem_raw + 16 * SSTR * 4);           // [16][SSTR]
  float*  Cts = (float*)(smem_raw + 16 * SSTR * 4 + 16 * SSTR * 2);

  const int bh = blockIdx.x;                 // b*16 + h
  const int q0 = blockIdx.y * 16;
  const bf16_t* Qbh = Q  + (size_t)bh * S_DIM * DKV;
  const bf16_t* Kbh = K  + (size_t)bh * S_DIM * DKV;
  const bf16_t* Vbh = Vt + (size_t)bh * DKV * S_DIM;

  const int tid = threadIdx.x;
  const int wave = tid >> 5;
  const int lane = tid & 31;
  const int col = lane & 15;
  const int rbase = (lane >> 4) * 8;

  // Q tile A-fragments (16x64 = two 16x32 K chunks), direct from HBM/L2
  v16bf qa0 = load_a_frag(Qbh + (size_t)q0 * DKV +  0, DKV);
  v16bf qa1 = load_a_frag(Qbh + (size_t)q0 * DKV + 32, DKV);

  // ---- scores: S = Q K^T, 128 k-tiles of 16 columns, 16 per wave ----------
  for (int kt = wave; kt < S_DIM / 16; kt += 8) {
    if (kt + 8 < S_DIM / 16)
      __builtin_prefetch(Kbh + (size_t)((kt + 8) * 16) * DKV, 0, 0);
    v8f acc = {};
    v16bf b0 = load_b_frag(Kbh + (size_t)(kt * 16) * DKV +  0, DKV);
    v16bf b1 = load_b_frag(Kbh + (size_t)(kt * 16) * DKV + 32, DKV);
    acc = wmma_bf16(qa0, b0, acc);
    acc = wmma_bf16(qa1, b1, acc);
#pragma unroll
    for (int g = 0; g < 8; ++g)
      Ssc[(rbase + g) * SSTR + kt * 16 + col] = acc[g];
  }
  __syncthreads();

  // ---- row softmax (scale 1/sqrt(64)=0.125), write attn, quantize P ------
  for (int rr = 0; rr < 2; ++rr) {
    int r = wave * 2 + rr;
    float* row = &Ssc[r * SSTR];
    float m = -3.0e38f;
    for (int i = lane; i < S_DIM; i += 32) m = fmaxf(m, row[i]);
#pragma unroll
    for (int off = 16; off > 0; off >>= 1) m = fmaxf(m, __shfl_xor(m, off, 32));
    float sum = 0.f;
    for (int i = lane; i < S_DIM; i += 32) {
      float e = __expf(0.125f * (row[i] - m));
      row[i] = e;
      sum += e;
    }
#pragma unroll
    for (int off = 16; off > 0; off >>= 1) sum += __shfl_xor(sum, off, 32);
    float inv = 1.0f / sum;
    float* arow = attn_out + ((size_t)bh * S_DIM + q0 + r) * S_DIM;
    for (int i = lane; i < S_DIM; i += 32) {
      float p = row[i] * inv;
      arow[i] = p;                // materialized attention output
      Pb[r * SSTR + i] = f2bf(p); // bf16 P for the PV GEMM
    }
  }
  for (int i = tid; i < 16 * DKV; i += 256) Cts[i] = 0.f;
  __syncthreads();

  // ---- context: C = P(16x2048) @ V(2048x64); K chunks split across waves --
  v8f cacc[4] = {{}, {}, {}, {}};
  for (int kc = wave; kc < S_DIM / 32; kc += 8) {
    v16bf pa = load_a_frag(&Pb[kc * 32], SSTR);
#pragma unroll
    for (int nt = 0; nt < 4; ++nt) {
      v16bf vb = load_b_frag(Vbh + (size_t)(nt * 16) * S_DIM + kc * 32, S_DIM);
      cacc[nt] = wmma_bf16(pa, vb, cacc[nt]);
    }
  }
  // cross-wave reduction via LDS float atomics (ds_add_f32)
#pragma unroll
  for (int nt = 0; nt < 4; ++nt)
#pragma unroll
    for (int g = 0; g < 8; ++g)
      atomicAdd(&Cts[(rbase + g) * DKV + nt * 16 + col], cacc[nt][g]);
  __syncthreads();

  const int b = bh >> 4, h = bh & 15;
  for (int i = tid; i < 16 * DKV; i += 256) {
    int r = i >> 6, d = i & 63;
    ctx_out[((size_t)(b * S_DIM + q0 + r)) * (N_HEAD * DKV) + h * DKV + d] =
        Cts[i];
  }
}

// ---------------------------------------------------------------------------
extern "C" void kernel_launch(void* const* d_in, const int* in_sizes, int n_in,
                              void* d_out, int out_size, void* d_ws,
                              size_t ws_size, hipStream_t stream) {
  const float* Xq = (const float*)d_in[0];
  const float* Xk = (const float*)d_in[1];
  const float* Xv = (const float*)d_in[2];
  const float* Wq = (const float*)d_in[3];
  const float* Wk = (const float*)d_in[4];
  const float* Wv = (const float*)d_in[5];

  float* ctx  = (float*)d_out;                               // B*S*1024 fp32
  float* attn = ctx + (size_t)B_DIM * S_DIM * N_HEAD * DKV;  // B*H*S*S fp32

  const size_t nproj = (size_t)B_DIM * S_DIM * N_HEAD * DKV; // 8.4M elems
  const size_t nw    = (size_t)D_MODEL * D_MODEL;            // 1M elems
  bf16_t* Qw  = (bf16_t*)d_ws;
  bf16_t* Kw  = Qw + nproj;
  bf16_t* Vw  = Kw + nproj;
  bf16_t* Xbq = Vw + nproj;
  bf16_t* Xbk = Xbq + nproj;
  bf16_t* Xbv = Xbk + nproj;
  bf16_t* Wtq = Xbv + nproj;
  bf16_t* Wtk = Wtq + nw;
  bf16_t* Wtv = Wtk + nw;

  // Prepass: bf16 conversion (X elementwise; W converted + transposed)
  dim3 gx(nproj / (256 * 4), 3);
  convert_x_kernel<<<gx, 256, 0, stream>>>(Xq, Xk, Xv, Xbq, Xbk, Xbv);
  dim3 gw(D_MODEL / 32, D_MODEL / 32, 3);
  convert_w_kernel<<<gw, 256, 0, stream>>>(Wq, Wk, Wv, Wtq, Wtk, Wtv);

  // Stage 1: QKV projections (z: 0=Q, 1=K, 2=V-transposed)
  dim3 g1((B_DIM * S_DIM) / GBM, (N_HEAD * DKV) / GBN, 3);
  qkv_proj_kernel<<<g1, 256, 0, stream>>>(Xbq, Xbk, Xbv, Wtq, Wtk, Wtv,
                                          Qw, Kw, Vw);

  // Stage 2: attention; >64KB dynamic LDS (gfx1250: up to 320KB per WG)
  (void)hipFuncSetAttribute((const void*)attention_kernel,
                            hipFuncAttributeMaxDynamicSharedMemorySize,
                            ATTN_SMEM_BYTES);
  dim3 g2(B_DIM * N_HEAD, S_DIM / 16);
  attention_kernel<<<g2, 256, ATTN_SMEM_BYTES, stream>>>(Qw, Kw, Vw, ctx, attn);
}
